// PhaseEncodingSNN_32547262169284
// MI455X (gfx1250) — compile-verified, matrix operand order checked by reading, not run on hardware
//
#include <hip/hip_runtime.h>
#include <hip/hip_bf16.h>

// ---------------------------------------------------------------------------
// PhaseEncodingSNN on MI455X (gfx1250): f16 WMMA pipeline.
//   B=64, T=512, I=256, H=512, O=256, BETA=0.9, THR=1.0
// Recurrent phase runs as ONE persistent kernel (4 WGPs, grid barrier per
// step, W_hh slice resident in LDS) instead of 512 sequential launches.
// ---------------------------------------------------------------------------

#define SNN_B 64
#define SNN_T 512
#define SNN_I 256
#define SNN_H 512
#define SNN_O 256
#define REC_BLOCKS 4
#define LDS_PITCH 520   // 512 + 8 halves pad -> bank stride 4 dwords, conflict-free

typedef __attribute__((ext_vector_type(16))) _Float16 v16h;
typedef __attribute__((ext_vector_type(8)))  _Float16 v8h;
typedef __attribute__((ext_vector_type(8)))  float    v8f;

// ---------------------------------------------------------------------------
// Fragment loader: 16x32 f16 tile in WMMA A/B register layout (ISA 7.12.2).
// For A: rows are M.  For B (K x N logical): B(k,n) = W[n][k], so loading W
// row-major with "row = n" yields the B fragment directly.
// Lane L: row = L&15 ; K-base = 8*(L>>4); halves 0..7 = K kb..kb+7,
// halves 8..15 = K 16+kb..16+kb+7.   Two 16-byte loads per lane.
// ---------------------------------------------------------------------------
__device__ __forceinline__ v16h frag_load16x32(const _Float16* __restrict__ base,
                                               int ld, int r0, int kk, int lane) {
    const int row = r0 + (lane & 15);
    const int kb  = kk + ((lane >> 4) << 3);
    const _Float16* p = base + (size_t)row * ld + kb;
    v8h lo = *reinterpret_cast<const v8h*>(p);
    v8h hi = *reinterpret_cast<const v8h*>(p + 16);
    v16h out;
#pragma unroll
    for (int j = 0; j < 8; ++j) { out[j] = lo[j]; out[8 + j] = hi[j]; }
    return out;
}

// Same fragment layout but sourced from an LDS-resident, row-padded tile.
__device__ __forceinline__ v16h frag_lds16x32(const _Float16* lds, int r0, int kk, int lane) {
    const int row = r0 + (lane & 15);
    const int kb  = kk + ((lane >> 4) << 3);
    const _Float16* p = lds + row * LDS_PITCH + kb;
    v8h lo = *reinterpret_cast<const v8h*>(p);
    v8h hi = *reinterpret_cast<const v8h*>(p + 16);
    v16h out;
#pragma unroll
    for (int j = 0; j < 8; ++j) { out[j] = lo[j]; out[8 + j] = hi[j]; }
    return out;
}

// ---------------------------------------------------------------------------
// Generic f16 GEMM:  C[M,N] = A[M,K] * W[N,K]^T + bias[N]   (f32 accumulate)
// Block: 256 threads = 8 waves (2 x 4), block tile 64(M) x 128(N),
// each wave owns a 32x32 tile = 2x2 WMMA fragments.
// grid.x = N/128, grid.y = M/64.
// ---------------------------------------------------------------------------
__global__ __launch_bounds__(256)
void gemm_f16_wmma(const _Float16* __restrict__ A, int lda,
                   const _Float16* __restrict__ W, int ldw,
                   const float* __restrict__ bias,
                   float* __restrict__ C, int ldc, int K) {
    const int lane = threadIdx.x & 31;
    const int wave = threadIdx.x >> 5;
    const int m0 = blockIdx.y * 64  + (wave >> 2) * 32;
    const int n0 = blockIdx.x * 128 + (wave & 3) * 32;

    v8f c00 = {}, c01 = {}, c10 = {}, c11 = {};
    for (int kk = 0; kk < K; kk += 32) {
        v16h a0 = frag_load16x32(A, lda, m0,      kk, lane);
        v16h a1 = frag_load16x32(A, lda, m0 + 16, kk, lane);
        v16h b0 = frag_load16x32(W, ldw, n0,      kk, lane);
        v16h b1 = frag_load16x32(W, ldw, n0 + 16, kk, lane);
        c00 = __builtin_amdgcn_wmma_f32_16x16x32_f16(false, a0, false, b0, (short)0, c00, false, false);
        c01 = __builtin_amdgcn_wmma_f32_16x16x32_f16(false, a0, false, b1, (short)0, c01, false, false);
        c10 = __builtin_amdgcn_wmma_f32_16x16x32_f16(false, a1, false, b0, (short)0, c10, false, false);
        c11 = __builtin_amdgcn_wmma_f32_16x16x32_f16(false, a1, false, b1, (short)0, c11, false, false);
    }
    // C/D layout: n = lane&15 ; m = vgpr_index + 8*(lane>>4)
    const int cn   = lane & 15;
    const int mrow = (lane >> 4) * 8;
    const float bn0 = bias[n0 + cn];
    const float bn1 = bias[n0 + 16 + cn];
#pragma unroll
    for (int r = 0; r < 8; ++r) {
        const size_t rA = (size_t)(m0 + mrow + r) * ldc;
        const size_t rB = (size_t)(m0 + 16 + mrow + r) * ldc;
        C[rA + n0 + cn]      = c00[r] + bn0;
        C[rA + n0 + 16 + cn] = c01[r] + bn1;
        C[rB + n0 + cn]      = c10[r] + bn0;
        C[rB + n0 + 16 + cn] = c11[r] + bn1;
    }
}

// ---------------------------------------------------------------------------
// Persistent recurrent phase: 512 LIF-GEMM steps in one kernel.
//   total = in_proj[t] + mem_h @ W_hh^T + b_hh
//   new_mem = BETA*mem + total ; spike = new_mem > 1 ; mem' = new_mem - spike
// Grid: 4 blocks x 256 threads; block b owns output columns [b*128, b*128+128).
// Its W_hh slice (512x128 f16) is staged once into LDS (row-padded, bank-
// conflict-free ds_load_b128 fragment reads).  The f16 membrane ping-pongs
// through global memory; steps are separated by an atomic grid barrier.
// ---------------------------------------------------------------------------
__global__ __launch_bounds__(256)
void rec_persistent_wmma(const _Float16* __restrict__ Whh,   // f16 (H,H) [g][h]
                         const float*    __restrict__ b_hh,
                         const float*    __restrict__ inproj, // (T,B,H) f32
                         float*          __restrict__ mem,    // (B,H) f32 master
                         _Float16*       __restrict__ memh0,  // (B,H) f16 ping
                         _Float16*       __restrict__ memh1,  // (B,H) f16 pong
                         _Float16*       __restrict__ spikes, // (T,B,H) f16 out
                         unsigned*       __restrict__ bar_ctr)
{
    extern __shared__ _Float16 lds[];            // 128 rows x LDS_PITCH halves
    const int lane = threadIdx.x & 31;
    const int wave = threadIdx.x >> 5;
    const int nblk = blockIdx.x * 128;           // this block's column base

    // ---- stage W_hh slice into LDS once: rows nblk..nblk+127, K=512 ------
    for (int c = threadIdx.x; c < 128 * 64; c += 256) {     // 16B chunks
        const int r = c >> 6;
        const int j = (c & 63) << 3;
        *reinterpret_cast<v8h*>(&lds[r * LDS_PITCH + j]) =
            *reinterpret_cast<const v8h*>(Whh + ((size_t)(nblk + r) << 9) + j);
    }
    __syncthreads();

    const int m0 = (wave >> 2) * 32;             // M = 64 rows total
    const int nw = (wave & 3) * 32;              // column offset within block
    const int n0 = nblk + nw;                    // global column
    const int cn   = lane & 15;
    const int mrow = (lane >> 4) * 8;
    const float bn0 = b_hh[n0 + cn];
    const float bn1 = b_hh[n0 + 16 + cn];

    for (int t = 0; t < SNN_T; ++t) {
        const _Float16* memh_in  = (t & 1) ? memh1 : memh0;
        _Float16*       memh_out = (t & 1) ? memh0 : memh1;
        const float* ip_t = inproj + (size_t)t * SNN_B * SNN_H;
        _Float16*    spk_t = spikes + (size_t)t * SNN_B * SNN_H;

        v8f c00 = {}, c01 = {}, c10 = {}, c11 = {};
#pragma unroll 4
        for (int kk = 0; kk < SNN_H; kk += 32) {
            v16h a0 = frag_load16x32(memh_in, SNN_H, m0,      kk, lane);
            v16h a1 = frag_load16x32(memh_in, SNN_H, m0 + 16, kk, lane);
            v16h b0 = frag_lds16x32(lds, nw,      kk, lane);
            v16h b1 = frag_lds16x32(lds, nw + 16, kk, lane);
            c00 = __builtin_amdgcn_wmma_f32_16x16x32_f16(false, a0, false, b0, (short)0, c00, false, false);
            c01 = __builtin_amdgcn_wmma_f32_16x16x32_f16(false, a0, false, b1, (short)0, c01, false, false);
            c10 = __builtin_amdgcn_wmma_f32_16x16x32_f16(false, a1, false, b0, (short)0, c10, false, false);
            c11 = __builtin_amdgcn_wmma_f32_16x16x32_f16(false, a1, false, b1, (short)0, c11, false, false);
        }

        // ---- fused LIF epilogue on this block's columns ------------------
#pragma unroll
        for (int r = 0; r < 8; ++r) {
#pragma unroll
            for (int q = 0; q < 4; ++q) {
                const int m = m0 + ((q & 2) ? 16 : 0) + mrow + r;
                const int n = n0 + ((q & 1) ? 16 : 0) + cn;
                const float acc = (q == 0) ? c00[r] : (q == 1) ? c01[r]
                                : (q == 2) ? c10[r] : c11[r];
                const float bb  = (q & 1) ? bn1 : bn0;
                const size_t idx = (size_t)m * SNN_H + n;
                const float total = acc + bb + ip_t[idx];
                const float nm = 0.9f * mem[idx] + total;
                const float sp = (nm > 1.0f) ? 1.0f : 0.0f;
                const float mv = nm - sp;        // subtract-reset, THR = 1.0
                mem[idx]      = mv;
                memh_out[idx] = (_Float16)mv;
                spk_t[idx]    = (_Float16)sp;
            }
        }

        // ---- grid barrier: release writes, arrive, acquire-spin ----------
        __threadfence();
        __syncthreads();
        if (threadIdx.x == 0) {
            __hip_atomic_fetch_add(bar_ctr, 1u, __ATOMIC_ACQ_REL, __HIP_MEMORY_SCOPE_AGENT);
            const unsigned target = (unsigned)(t + 1) * REC_BLOCKS;
            while (__hip_atomic_load(bar_ctr, __ATOMIC_ACQUIRE, __HIP_MEMORY_SCOPE_AGENT) < target) {
                __builtin_amdgcn_s_sleep(2);
            }
        }
        __syncthreads();
    }
}

// ---------------------------------------------------------------------------
// LIF scan over T for layer 1; also transposes (B,T,H) -> (T,B,H) f16 spikes.
// ---------------------------------------------------------------------------
__global__ __launch_bounds__(256)
void lif_scan1(const float* __restrict__ iproj,  // (B,T,H)
               _Float16* __restrict__ spk)       // (T,B,H)
{
    const int tid = blockIdx.x * blockDim.x + threadIdx.x;  // over B*H
    const int b = tid >> 9;         // /H
    const int h = tid & (SNN_H - 1);
    const float* src = iproj + (size_t)b * SNN_T * SNN_H + h;
    _Float16*    dst = spk   + (size_t)b * SNN_H + h;
    float m = 0.0f;
    for (int t = 0; t < SNN_T; ++t) {
        const float nm = 0.9f * m + src[(size_t)t * SNN_H];
        const float sp = (nm > 1.0f) ? 1.0f : 0.0f;
        m = nm - sp;
        dst[(size_t)t * SNN_B * SNN_H] = (_Float16)sp;
    }
}

// ---------------------------------------------------------------------------
// Softmax over T per output channel o.
// ---------------------------------------------------------------------------
__global__ void softmax_t(const float* __restrict__ ta, float* __restrict__ attn) {
    const int o = blockIdx.x * blockDim.x + threadIdx.x;
    if (o >= SNN_O) return;
    float mx = -3.4e38f;
    for (int t = 0; t < SNN_T; ++t) mx = fmaxf(mx, ta[t * SNN_O + o]);
    float s = 0.0f;
    for (int t = 0; t < SNN_T; ++t) s += __expf(ta[t * SNN_O + o] - mx);
    const float inv = 1.0f / s;
    for (int t = 0; t < SNN_T; ++t)
        attn[t * SNN_O + o] = __expf(ta[t * SNN_O + o] - mx) * inv;
}

// ---------------------------------------------------------------------------
// attended[b,o] = sum_t attn[t,o] * out_proj[t,b,o]; final spike threshold.
// ---------------------------------------------------------------------------
__global__ __launch_bounds__(256)
void attend_threshold(const float* __restrict__ oproj,  // (T,B,O)
                      const float* __restrict__ attn,   // (T,O)
                      float* __restrict__ out)          // (B,O)
{
    const int tid = blockIdx.x * blockDim.x + threadIdx.x;  // B*O
    const int b = tid >> 8;
    const int o = tid & (SNN_O - 1);
    float acc = 0.0f;
    for (int t = 0; t < SNN_T; ++t)
        acc += attn[t * SNN_O + o] * oproj[((size_t)t * SNN_B + b) * SNN_O + o];
    out[tid] = (acc > 1.0f) ? 1.0f : 0.0f;
}

// ---------------------------------------------------------------------------
__global__ void cvt_f32_f16(const float* __restrict__ in, _Float16* __restrict__ out, int n) {
    const int i = blockIdx.x * blockDim.x + threadIdx.x;
    if (i < n) out[i] = (_Float16)in[i];
}

__global__ void init_state(float* __restrict__ mem, _Float16* __restrict__ memh0,
                           unsigned* __restrict__ bar_ctr) {
    const int i = blockIdx.x * blockDim.x + threadIdx.x;
    if (i < SNN_B * SNN_H) { mem[i] = 0.0f; memh0[i] = (_Float16)0.0f; }
    if (i == 0) *bar_ctr = 0u;
}

// ---------------------------------------------------------------------------
extern "C" void kernel_launch(void* const* d_in, const int* in_sizes, int n_in,
                              void* d_out, int out_size, void* d_ws, size_t ws_size,
                              hipStream_t stream) {
    const float* x      = (const float*)d_in[0];   // (B,T,I)
    const float* W_in   = (const float*)d_in[1];   // (H,I)
    const float* b_in   = (const float*)d_in[2];   // (H)
    const float* W_ih   = (const float*)d_in[3];   // (H,H)
    const float* b_ih   = (const float*)d_in[4];   // (H)
    const float* W_hh   = (const float*)d_in[5];   // (H,H)
    const float* b_hh   = (const float*)d_in[6];   // (H)
    const float* W_out  = (const float*)d_in[7];   // (O,H)
    const float* b_out  = (const float*)d_in[8];   // (O)
    const float* t_attn = (const float*)d_in[9];   // (T,O)
    float* out = (float*)d_out;

    char* ws = (char*)d_ws;
    // -------- workspace layout (out_proj overlays dead input_proj) ---------
    const size_t XH_OFF     = 0;                                         // 16 MB f16
    const size_t IPROJ_OFF  = XH_OFF + (size_t)SNN_B*SNN_T*SNN_I*2;      // 64 MB f32
    const size_t SPK_OFF    = IPROJ_OFF + (size_t)SNN_B*SNN_T*SNN_H*4;   // 32 MB f16
    const size_t INPROJ_OFF = SPK_OFF + (size_t)SNN_T*SNN_B*SNN_H*2;     // 64 MB f32
    const size_t TAIL       = INPROJ_OFF + (size_t)SNN_T*SNN_B*SNN_H*4;
    const size_t OPROJ_OFF  = IPROJ_OFF;                                 // reuse: 32 MB
    size_t off = TAIL;
    const size_t WIN_OFF  = off; off += (size_t)SNN_H*SNN_I*2;
    const size_t WIH_OFF  = off; off += (size_t)SNN_H*SNN_H*2;
    const size_t WHH_OFF  = off; off += (size_t)SNN_H*SNN_H*2;
    const size_t WOUT_OFF = off; off += (size_t)SNN_O*SNN_H*2;
    const size_t MEM_OFF  = off; off += (size_t)SNN_B*SNN_H*4;
    const size_t MEMH0_OFF= off; off += (size_t)SNN_B*SNN_H*2;
    const size_t MEMH1_OFF= off; off += (size_t)SNN_B*SNN_H*2;
    const size_t ATTN_OFF = off; off += (size_t)SNN_T*SNN_O*4;
    const size_t CTR_OFF  = off; off += 256;

    _Float16* xh     = (_Float16*)(ws + XH_OFF);
    float*    iproj  = (float*)   (ws + IPROJ_OFF);
    _Float16* spikes = (_Float16*)(ws + SPK_OFF);
    float*    inproj = (float*)   (ws + INPROJ_OFF);
    float*    oproj  = (float*)   (ws + OPROJ_OFF);
    _Float16* win_h  = (_Float16*)(ws + WIN_OFF);
    _Float16* wih_h  = (_Float16*)(ws + WIH_OFF);
    _Float16* whh_h  = (_Float16*)(ws + WHH_OFF);
    _Float16* wout_h = (_Float16*)(ws + WOUT_OFF);
    float*    mem    = (float*)   (ws + MEM_OFF);
    _Float16* memh0  = (_Float16*)(ws + MEMH0_OFF);
    _Float16* memh1  = (_Float16*)(ws + MEMH1_OFF);
    float*    attn   = (float*)   (ws + ATTN_OFF);
    unsigned* barctr = (unsigned*)(ws + CTR_OFF);

    // -------- precision conversion (weights hot in L2 afterwards) ---------
    {
        const int n = SNN_B*SNN_T*SNN_I;
        cvt_f32_f16<<<(n + 255) / 256, 256, 0, stream>>>(x, xh, n);
    }
    cvt_f32_f16<<<(SNN_H*SNN_I + 255)/256, 256, 0, stream>>>(W_in,  win_h,  SNN_H*SNN_I);
    cvt_f32_f16<<<(SNN_H*SNN_H + 255)/256, 256, 0, stream>>>(W_ih,  wih_h,  SNN_H*SNN_H);
    cvt_f32_f16<<<(SNN_H*SNN_H + 255)/256, 256, 0, stream>>>(W_hh,  whh_h,  SNN_H*SNN_H);
    cvt_f32_f16<<<(SNN_O*SNN_H + 255)/256, 256, 0, stream>>>(W_out, wout_h, SNN_O*SNN_H);

    // -------- phase 1: input projection (B*T,I)x(I,H) ---------------------
    gemm_f16_wmma<<<dim3(SNN_H/128, (SNN_B*SNN_T)/64), 256, 0, stream>>>(
        xh, SNN_I, win_h, SNN_I, b_in, iproj, SNN_H, SNN_I);

    // -------- phase 2: LIF scan 1 (+ transpose to (T,B,H) f16 spikes) -----
    lif_scan1<<<(SNN_B*SNN_H)/256, 256, 0, stream>>>(iproj, spikes);

    // -------- phase 3: in_proj GEMM on spikes ------------------------------
    gemm_f16_wmma<<<dim3(SNN_H/128, (SNN_T*SNN_B)/64), 256, 0, stream>>>(
        spikes, SNN_H, wih_h, SNN_H, b_ih, inproj, SNN_H, SNN_H);

    // -------- phase 4: persistent recurrent scan (512 fused steps) ---------
    init_state<<<(SNN_B*SNN_H)/256, 256, 0, stream>>>(mem, memh0, barctr);
    {
        const size_t lds_bytes = (size_t)128 * LDS_PITCH * sizeof(_Float16); // 133,120 B
        rec_persistent_wmma<<<dim3(REC_BLOCKS), 256, lds_bytes, stream>>>(
            whh_h, b_hh, inproj, mem, memh0, memh1, spikes, barctr);
    }

    // -------- phase 5: output projection ----------------------------------
    gemm_f16_wmma<<<dim3(SNN_O/128, (SNN_T*SNN_B)/64), 256, 0, stream>>>(
        spikes, SNN_H, wout_h, SNN_H, b_out, oproj, SNN_O, SNN_H);

    // -------- phase 6: temporal softmax attention + final threshold -------
    softmax_t<<<1, SNN_O, 0, stream>>>(t_attn, attn);
    attend_threshold<<<(SNN_B*SNN_O)/256, 256, 0, stream>>>(oproj, attn, out);

    (void)in_sizes; (void)n_in; (void)out_size; (void)ws_size;
}
